// E3_transformer_42528766165476
// MI455X (gfx1250) — matrix operand order repaired and checked by reading, not compiled, not run on hardware
//
#include <hip/hip_runtime.h>

// ---------------------------------------------------------------------------
// E3 transformer layer for MI455X (gfx1250, wave32, WMMA).
// Dominant cost: per-edge LN-MLP (128->128 relu ->432) x2 paths = 17.6 GFLOP
// -> v_wmma_f32_16x16x32_f16, 140 WMMAs per 16-edge tile per path.
// Dead code eliminated: W111/ve path, dot_w1e, hh_w1e never reach outputs.
// edge_dst is sorted with exactly K=30 edges per node -> no atomics.
// ---------------------------------------------------------------------------

typedef _Float16 v16h __attribute__((ext_vector_type(16)));
typedef _Float16 v8h  __attribute__((ext_vector_type(8)));
typedef _Float16 v4h  __attribute__((ext_vector_type(4)));
typedef float    v8f  __attribute__((ext_vector_type(8)));

#define NB   4
#define LLEN 512
#define NN   2048           // NB*LLEN nodes
#define KE   30             // edges per destination node
#define EE   61440          // NB*LLEN*KE edges
#define DP   128            // D_PAIR
#define WN   432            // W_NUMEL
#define NTILES (EE/16)      // 3840 edge tiles

#define INV_SQRT2 0.70710678f
#define INV_SQRT3 0.57735027f
#define INV_SQRT5 0.44721360f
#define INV_SQRT6 0.40824829f
#define A0_C   0.22360680f   // sqrt(1/20)
#define A1O_C  0.35355339f   // sqrt(3/24)
#define ADOT_C 0.058925565f  // sqrt(1/288)
#define INV_DH 0.15811388f   // 1/sqrt(16+6*4)

// ---------------- weight prep: f32 [R][C] -> f16 transposed [C][R] ----------
__global__ void wprep_kernel(const float* __restrict__ src,
                             _Float16* __restrict__ dst, int R, int C) {
  int i = blockIdx.x * blockDim.x + threadIdx.x;
  if (i >= R * C) return;
  int r = i / C, c = i - r * C;
  dst[c * R + r] = (_Float16)src[i];
}

// ---------------- node prep: s, folded queries qd, qvd ----------------------
__global__ void node_prep_kernel(const float* __restrict__ node,
                                 const float* __restrict__ l1,
                                 const float* __restrict__ pw,   // [256][16]
                                 const float* __restrict__ pb,   // [16]
                                 const float* __restrict__ hq0,  // [16][16]
                                 const float* __restrict__ hq1,  // [4][4]
                                 const float* __restrict__ dw0,  // [16][16]
                                 const float* __restrict__ dw1o, // [4][4]
                                 float* __restrict__ s_out,      // [NN][16]
                                 float* __restrict__ qd_out,     // [NN][16]
                                 float* __restrict__ qvd_out) {  // [NN][12]
  int n = blockIdx.x * blockDim.x + threadIdx.x;
  if (n >= NN) return;
  const float* x = node + (long)n * 256;
  float sv[16];
#pragma unroll
  for (int o = 0; o < 16; ++o) sv[o] = pb[o];
  for (int c = 0; c < 256; ++c) {
    float xc = x[c];
    const float* wr = pw + c * 16;
#pragma unroll
    for (int o = 0; o < 16; ++o) sv[o] += xc * wr[o];
  }
#pragma unroll
  for (int o = 0; o < 16; ++o) s_out[n * 16 + o] = sv[o];
  // qs = s @ hq_w0 / 4 ;  qd = qs @ dot_w0
  float qs[16];
#pragma unroll
  for (int v = 0; v < 16; ++v) {
    float a = 0.f;
#pragma unroll
    for (int u = 0; u < 16; ++u) a += sv[u] * hq0[u * 16 + v];
    qs[v] = 0.25f * a;
  }
#pragma unroll
  for (int w = 0; w < 16; ++w) {
    float a = 0.f;
#pragma unroll
    for (int u = 0; u < 16; ++u) a += qs[u] * dw0[u * 16 + w];
    qd_out[n * 16 + w] = a;
  }
  // qvo = einsum(v, hq_w1)/2 ; qvd = einsum(qvo, dot_w1o)
  float vf[4][3], qv[4][3];
#pragma unroll
  for (int u = 0; u < 4; ++u)
#pragma unroll
    for (int i = 0; i < 3; ++i) vf[u][i] = l1[n * 12 + u * 3 + i];
#pragma unroll
  for (int v = 0; v < 4; ++v)
#pragma unroll
    for (int i = 0; i < 3; ++i) {
      float a = 0.f;
#pragma unroll
      for (int u = 0; u < 4; ++u) a += vf[u][i] * hq1[u * 4 + v];
      qv[v][i] = 0.5f * a;
    }
#pragma unroll
  for (int w = 0; w < 4; ++w)
#pragma unroll
    for (int i = 0; i < 3; ++i) {
      float a = 0.f;
#pragma unroll
      for (int u = 0; u < 4; ++u) a += qv[u][i] * dw1o[u * 4 + w];
      qvd_out[n * 12 + w * 3 + i] = a;
    }
}

// ---------------- edge kernel: gather + LN + WMMA MLP + tensor product ------
// One wave per 16-edge tile. LDS: A(16x128 f16) | H(16x128 f16) | OUT(16x432 f32)
__global__ void __launch_bounds__(32)
edge_mlp_tp_kernel(const float* __restrict__ pair,
                   const float* __restrict__ s,       // [NN][16]
                   const float* __restrict__ l1,      // [NN][12] (= v)
                   const float* __restrict__ sh,      // [EE][9]
                   const int*   __restrict__ pj,      // pair_index[2]
                   const int*   __restrict__ esrc,
                   const int*   __restrict__ edst,
                   const float* __restrict__ ln_g,
                   const float* __restrict__ ln_b,
                   const float* __restrict__ b1,
                   const float* __restrict__ b2,
                   const _Float16* __restrict__ w1T,  // [128][128] f16 (out-major)
                   const _Float16* __restrict__ w2T,  // [432][128] f16 (out-major)
                   const float* __restrict__ qd,      // [NN][16]
                   const float* __restrict__ qvd,     // [NN][12]
                   float* __restrict__ expv,          // [EE]      (mode 0)
                   float* __restrict__ vflat,         // [EE][28]  (mode 1)
                   int mode) {
  extern __shared__ char smem[];
  _Float16* ldsA = (_Float16*)smem;            // 16*128 f16 = 4 KB
  _Float16* ldsH = (_Float16*)(smem + 4096);   // 16*128 f16 = 4 KB
  float*    ldsO = (float*)(smem + 8192);      // 16*432 f32 = 27648 B
  const int lane = threadIdx.x & 31;
  const int e0 = blockIdx.x * 16;

  // ---- gather pair rows + LayerNorm into ldsA (A-matrix staging) ----
  int rowBase = 0;
  if (lane < 16) {
    int e = e0 + lane;
    rowBase = (edst[e] * LLEN + pj[e]) * DP;   // < 2^28, fits in int
  }
  float4 g4 = ((const float4*)ln_g)[lane];
  float4 bb4 = ((const float4*)ln_b)[lane];
  for (int r = 0; r < 16; ++r) {
    int rb = __shfl(rowBase, r);
    float4 x = ((const float4*)(pair + (long)rb))[lane];
    float sm = x.x + x.y + x.z + x.w;
    float sq = x.x * x.x + x.y * x.y + x.z * x.z + x.w * x.w;
#pragma unroll
    for (int off = 16; off > 0; off >>= 1) {
      sm += __shfl_xor(sm, off);
      sq += __shfl_xor(sq, off);
    }
    float mu = sm * (1.0f / 128.0f);
    float var = sq * (1.0f / 128.0f) - mu * mu;
    float rs = rsqrtf(var + 1e-5f);
    v4h h;
    h[0] = (_Float16)((x.x - mu) * rs * g4.x + bb4.x);
    h[1] = (_Float16)((x.y - mu) * rs * g4.y + bb4.y);
    h[2] = (_Float16)((x.z - mu) * rs * g4.z + bb4.z);
    h[3] = (_Float16)((x.w - mu) * rs * g4.w + bb4.w);
    *(v4h*)&ldsA[r * DP + lane * 4] = h;
  }
  asm volatile("s_wait_dscnt 0" ::: "memory");

  // ---- load A fragments (ISA 16-bit A 16x32 layout), keep in VGPRs ----
  const int m   = lane & 15;
  const int khi = (lane >> 4) ? 8 : 0;   // A: lanes 16-31 hold K 8..15 / 24..31
  const int nlo = lane & 15;
  const int kbB = (lane >> 4) * 8;       // B: lanes 16-31 hold K 8..15 / 24..31
  v16h aF[4];
#pragma unroll
  for (int c = 0; c < 4; ++c) {
    const _Float16* p = &ldsA[m * DP + c * 32 + khi];
    v8h lo = *(const v8h*)p;
    v8h hi = *(const v8h*)(p + 16);
    aF[c] = __builtin_shufflevector(lo, hi, 0, 1, 2, 3, 4, 5, 6, 7, 8, 9, 10,
                                    11, 12, 13, 14, 15);
  }

  // ---- layer 1: (16x128) @ (128x128), bias + relu -> ldsH ----
#pragma unroll
  for (int t = 0; t < 8; ++t) {
    v8f acc = {};
    const _Float16* bbase = w1T + (t * 16 + nlo) * DP + kbB;
#pragma unroll
    for (int c = 0; c < 4; ++c) {
      v8h lo = *(const v8h*)(bbase + c * 32);
      v8h hi = *(const v8h*)(bbase + c * 32 + 16);
      v16h bF = __builtin_shufflevector(lo, hi, 0, 1, 2, 3, 4, 5, 6, 7, 8, 9,
                                        10, 11, 12, 13, 14, 15);
      acc = __builtin_amdgcn_wmma_f32_16x16x32_f16(false, aF[c], false, bF,
                                                   (short)0, acc, false, false);
    }
    float bias = b1[t * 16 + nlo];
#pragma unroll
    for (int i = 0; i < 8; ++i) {
      float hv = acc[i] + bias;
      hv = hv > 0.f ? hv : 0.f;
      int row = (lane < 16) ? i : (i + 8);
      ldsH[row * DP + t * 16 + nlo] = (_Float16)hv;
    }
  }
  asm volatile("s_wait_dscnt 0" ::: "memory");

  v16h hF[4];
#pragma unroll
  for (int c = 0; c < 4; ++c) {
    const _Float16* p = &ldsH[m * DP + c * 32 + khi];
    v8h lo = *(const v8h*)p;
    v8h hi = *(const v8h*)(p + 16);
    hF[c] = __builtin_shufflevector(lo, hi, 0, 1, 2, 3, 4, 5, 6, 7, 8, 9, 10,
                                    11, 12, 13, 14, 15);
  }

  // ---- layer 2: (16x128) @ (128x432), bias -> ldsO ----
#pragma unroll 3
  for (int t2 = 0; t2 < 27; ++t2) {
    v8f acc = {};
    const _Float16* bbase = w2T + (t2 * 16 + nlo) * DP + kbB;
#pragma unroll
    for (int c = 0; c < 4; ++c) {
      v8h lo = *(const v8h*)(bbase + c * 32);
      v8h hi = *(const v8h*)(bbase + c * 32 + 16);
      v16h bF = __builtin_shufflevector(lo, hi, 0, 1, 2, 3, 4, 5, 6, 7, 8, 9,
                                        10, 11, 12, 13, 14, 15);
      acc = __builtin_amdgcn_wmma_f32_16x16x32_f16(false, hF[c], false, bF,
                                                   (short)0, acc, false, false);
    }
    float bias = b2[t2 * 16 + nlo];
#pragma unroll
    for (int i = 0; i < 8; ++i) {
      int row = (lane < 16) ? i : (i + 8);
      ldsO[row * WN + t2 * 16 + nlo] = acc[i] + bias;
    }
  }
  asm volatile("s_wait_dscnt 0" ::: "memory");

  // ---- per-edge tensor product (lane = edge; W111/ve path is dead code) ----
  if (lane < 16) {
    int e = e0 + lane;
    const float* w = &ldsO[lane * WN];
    int src = esrc[e];
    const float* ssp = s + src * 16;
    const float* vp = l1 + src * 12;
    const float* shp = sh + (long)e * 9;
    float ssv[16];
#pragma unroll
    for (int u = 0; u < 16; ++u) ssv[u] = ssp[u];
    float vs[4][3];
#pragma unroll
    for (int u = 0; u < 4; ++u)
#pragma unroll
      for (int i = 0; i < 3; ++i) vs[u][i] = vp[u * 3 + i];
    float y0 = shp[0];
    float y1[3] = {shp[1], shp[2], shp[3]};
    float y2[5] = {shp[4], shp[5], shp[6], shp[7], shp[8]};

    float du[4];
#pragma unroll
    for (int u = 0; u < 4; ++u)
      du[u] = vs[u][0] * y1[0] + vs[u][1] * y1[1] + vs[u][2] * y1[2];

    float s_out[16];
#pragma unroll
    for (int o = 0; o < 16; ++o) {
      float a0 = 0.f, a1 = 0.f;
#pragma unroll
      for (int u = 0; u < 16; ++u) a0 += ssv[u] * w[u * 16 + o];      // W000
#pragma unroll
      for (int u = 0; u < 4; ++u) a1 += du[u] * w[256 + u * 16 + o];  // W110
      s_out[o] = A0_C * (y0 * a0 + INV_SQRT3 * a1);
    }

    // M[i][k] = sum_m y2[m] * T[m][i][k]  (then /sqrt5 in A121)
    float M[3][3];
    M[0][0] = -y2[2] * INV_SQRT6 + y2[4] * INV_SQRT2;
    M[0][1] = y2[0] * INV_SQRT2;
    M[0][2] = y2[3] * INV_SQRT2;
    M[1][0] = y2[0] * INV_SQRT2;
    M[1][1] = -y2[2] * INV_SQRT6 - y2[4] * INV_SQRT2;
    M[1][2] = y2[1] * INV_SQRT2;
    M[2][0] = y2[3] * INV_SQRT2;
    M[2][1] = y2[1] * INV_SQRT2;
    M[2][2] = 2.f * y2[2] * INV_SQRT6;
    float A121[4][3];
#pragma unroll
    for (int u = 0; u < 4; ++u)
#pragma unroll
      for (int k = 0; k < 3; ++k)
        A121[u][k] = INV_SQRT5 * (vs[u][0] * M[0][k] + vs[u][1] * M[1][k] +
                                  vs[u][2] * M[2][k]);

    float t1[4];
#pragma unroll
    for (int o = 0; o < 4; ++o) {
      float a = 0.f;
#pragma unroll
      for (int u = 0; u < 16; ++u) a += ssv[u] * w[320 + u * 4 + o];  // W011
      t1[o] = a;
    }
    float vo[4][3];
#pragma unroll
    for (int o = 0; o < 4; ++o)
#pragma unroll
      for (int k = 0; k < 3; ++k) {
        float term1 = INV_SQRT3 * t1[o] * y1[k];
        float term2 = 0.f, term3 = 0.f;
#pragma unroll
        for (int u = 0; u < 4; ++u) {
          term2 += vs[u][k] * w[384 + u * 4 + o];   // W101
          term3 += A121[u][k] * w[400 + u * 4 + o]; // W121
        }
        vo[o][k] = A1O_C * (term1 + INV_SQRT3 * y0 * term2 + term3);
      }

    if (mode == 0) {  // k path -> attention logits
      int d = edst[e];
      const float* qdp = qd + d * 16;
      const float* qvp = qvd + d * 12;
      float dot = 0.f;
#pragma unroll
      for (int o = 0; o < 16; ++o) dot += qdp[o] * s_out[o];
      float dv = 0.f;
#pragma unroll
      for (int o = 0; o < 4; ++o)
#pragma unroll
        for (int k = 0; k < 3; ++k) dv += qvp[o * 3 + k] * vo[o][k];
      dot = ADOT_C * (dot + dv * INV_SQRT3);
      expv[e] = __expf(dot * INV_DH);
    } else {          // v path -> message values
      float* out = vflat + (long)e * 28;
#pragma unroll
      for (int o = 0; o < 16; ++o) out[o] = s_out[o];
#pragma unroll
      for (int o = 0; o < 4; ++o)
#pragma unroll
        for (int k = 0; k < 3; ++k) out[16 + o * 3 + k] = vo[o][k];
    }
  }
}

// ---------------- node output: softmax-normalize, reduce 30 edges, heads ----
__global__ void node_out_kernel(const float* __restrict__ expv,
                                const float* __restrict__ vflat,
                                const float* __restrict__ hh0,   // [16][16]
                                const float* __restrict__ hh1o,  // [4][4]
                                const float* __restrict__ ho0,   // [16][16]
                                const float* __restrict__ ho1,   // [4][4]
                                const float* __restrict__ pnw,   // [16][256]
                                const float* __restrict__ pnb,   // [256]
                                const float* __restrict__ node,
                                const float* __restrict__ l1,
                                float* __restrict__ out_node,    // [NN][256]
                                float* __restrict__ out_l1) {    // [NN][12]
  int n = blockIdx.x * blockDim.x + threadIdx.x;
  if (n >= NN) return;
  const float* ev = expv + (long)n * KE;
  float z = 0.f;
  for (int k = 0; k < KE; ++k) z += ev[k];
  float inv = 1.f / (z + 1e-5f);
  float ms[28];
#pragma unroll
  for (int j = 0; j < 28; ++j) ms[j] = 0.f;
  for (int k = 0; k < KE; ++k) {
    float a = ev[k] * inv;
    const float* vf = vflat + ((long)n * KE + k) * 28;
#pragma unroll
    for (int j = 0; j < 28; ++j) ms[j] += a * vf[j];
  }
#pragma unroll
  for (int j = 0; j < 28; ++j) ms[j] *= (1.f / (float)KE);

  float os1[16], os2[16];
#pragma unroll
  for (int w = 0; w < 16; ++w) {
    float a = 0.f;
#pragma unroll
    for (int u = 0; u < 16; ++u) a += ms[u] * hh0[u * 16 + w];
    os1[w] = 0.25f * a;
  }
#pragma unroll
  for (int w = 0; w < 16; ++w) {
    float a = 0.f;
#pragma unroll
    for (int u = 0; u < 16; ++u) a += os1[u] * ho0[u * 16 + w];
    os2[w] = 0.25f * a;
  }
  float ov1[4][3], ov2[4][3];
#pragma unroll
  for (int w = 0; w < 4; ++w)
#pragma unroll
    for (int i = 0; i < 3; ++i) {
      float a = 0.f;
#pragma unroll
      for (int u = 0; u < 4; ++u) a += ms[16 + u * 3 + i] * hh1o[u * 4 + w];
      ov1[w][i] = 0.5f * a;
    }
#pragma unroll
  for (int w = 0; w < 4; ++w)
#pragma unroll
    for (int i = 0; i < 3; ++i) {
      float a = 0.f;
#pragma unroll
      for (int u = 0; u < 4; ++u) a += ov1[u][i] * ho1[u * 4 + w];
      ov2[w][i] = 0.5f * a;
    }
#pragma unroll
  for (int u = 0; u < 4; ++u)
#pragma unroll
    for (int i = 0; i < 3; ++i)
      out_l1[n * 12 + u * 3 + i] = ov2[u][i] + l1[n * 12 + u * 3 + i];

  const float* xn = node + (long)n * 256;
  for (int c = 0; c < 256; ++c) {
    float a = pnb[c] + xn[c];
#pragma unroll
    for (int w = 0; w < 16; ++w) a += os2[w] * pnw[w * 256 + c];
    out_node[n * 256 + c] = a;
  }
}

// ---------------------------------------------------------------------------
extern "C" void kernel_launch(void* const* d_in, const int* in_sizes, int n_in,
                              void* d_out, int out_size, void* d_ws,
                              size_t ws_size, hipStream_t stream) {
  // setup_inputs() order: node, pair, l1_feats, edge_sh, then params in dict
  // insertion order (26 tensors), then pair_index, edge_src, edge_dst.
  const float* node = (const float*)d_in[0];
  const float* pair = (const float*)d_in[1];
  const float* l1 = (const float*)d_in[2];
  const float* sh = (const float*)d_in[3];
  const float* proj_l0_w = (const float*)d_in[4];
  const float* proj_l0_b = (const float*)d_in[5];
  const float* hq_w0 = (const float*)d_in[6];
  const float* hq_w1 = (const float*)d_in[7];
  const float* k_ln_g = (const float*)d_in[8];
  const float* k_ln_b = (const float*)d_in[9];
  const float* k_w1 = (const float*)d_in[10];
  const float* k_b1 = (const float*)d_in[11];
  const float* k_w2 = (const float*)d_in[12];
  const float* k_b2 = (const float*)d_in[13];
  const float* v_ln_g = (const float*)d_in[14];
  const float* v_ln_b = (const float*)d_in[15];
  const float* v_w1 = (const float*)d_in[16];
  const float* v_b1 = (const float*)d_in[17];
  const float* v_w2 = (const float*)d_in[18];
  const float* v_b2 = (const float*)d_in[19];
  const float* dot_w0 = (const float*)d_in[20];
  const float* dot_w1o = (const float*)d_in[21];
  // d_in[22] = dot_w1e (dead), d_in[25] = hh_w1e (dead)
  const float* hh_w0 = (const float*)d_in[23];
  const float* hh_w1o = (const float*)d_in[24];
  const float* ho_w0 = (const float*)d_in[26];
  const float* ho_w1 = (const float*)d_in[27];
  const float* proj_node_w = (const float*)d_in[28];
  const float* proj_node_b = (const float*)d_in[29];
  const int* pidx = (const int*)d_in[30];
  const int* esrc = (const int*)d_in[31];
  const int* edst = (const int*)d_in[32];
  const int* pj = pidx + 2 * EE;

  // workspace carve (~7.8 MB total)
  float* ws_s = (float*)d_ws;               // NN*16
  float* ws_qd = ws_s + NN * 16;            // NN*16
  float* ws_qvd = ws_qd + NN * 16;          // NN*12
  float* ws_ex = ws_qvd + NN * 12;          // EE
  float* ws_vf = ws_ex + EE;                // EE*28
  _Float16* w1Tk = (_Float16*)(ws_vf + (size_t)EE * 28);
  _Float16* w2Tk = w1Tk + 128 * 128;
  _Float16* w1Tv = w2Tk + WN * 128;
  _Float16* w2Tv = w1Tv + 128 * 128;

  float* out_node = (float*)d_out;
  float* out_l1 = out_node + (size_t)NN * 256;

  // weight prep (tiny, L2-resident afterwards)
  wprep_kernel<<<(128 * 128 + 255) / 256, 256, 0, stream>>>(k_w1, w1Tk, 128, 128);
  wprep_kernel<<<(128 * WN + 255) / 256, 256, 0, stream>>>(k_w2, w2Tk, 128, WN);
  wprep_kernel<<<(128 * 128 + 255) / 256, 256, 0, stream>>>(v_w1, w1Tv, 128, 128);
  wprep_kernel<<<(128 * WN + 255) / 256, 256, 0, stream>>>(v_w2, w2Tv, 128, WN);

  node_prep_kernel<<<NN / 256, 256, 0, stream>>>(
      node, l1, proj_l0_w, proj_l0_b, hq_w0, hq_w1, dot_w0, dot_w1o, ws_s,
      ws_qd, ws_qvd);

  const size_t lds_bytes = 16 * DP * 2 * 2 + 16 * WN * 4;  // 35840 B (< 64 KB)
  edge_mlp_tp_kernel<<<NTILES, 32, lds_bytes, stream>>>(
      pair, ws_s, l1, sh, pj, esrc, edst, k_ln_g, k_ln_b, k_b1, k_b2, w1Tk,
      w2Tk, ws_qd, ws_qvd, ws_ex, ws_vf, 0);
  edge_mlp_tp_kernel<<<NTILES, 32, lds_bytes, stream>>>(
      pair, ws_s, l1, sh, pj, esrc, edst, v_ln_g, v_ln_b, v_b1, v_b2, w1Tv,
      w2Tv, ws_qd, ws_qvd, ws_ex, ws_vf, 1);

  node_out_kernel<<<NN / 256, 256, 0, stream>>>(
      ws_ex, ws_vf, hh_w0, hh_w1o, ho_w0, ho_w1, proj_node_w, proj_node_b,
      node, l1, out_node, out_l1);
}
